// ReaxFF_nn_force_40767829574061
// MI455X (gfx1250) — matrix operand order repaired and checked by reading, not compiled, not run on hardware
//
#include <hip/hip_runtime.h>
#include <hip/hip_bf16.h>
#include <math.h>

typedef __attribute__((ext_vector_type(16))) _Float16 v16h;
typedef __attribute__((ext_vector_type(8)))  float    v8f;

#define NBATCH 16
#define NATOM  768
#define NBOND  6144
#define NLAYER 14
#define TILES_PER_BATCH (NBOND / 16)               // 384
#define TILES_TOTAL     (NBATCH * TILES_PER_BATCH) // 6144

// fast sigmoid: v_exp_f32 + v_add + v_rcp_f32
__device__ __forceinline__ float sigmoidf_(float x) {
    return __builtin_amdgcn_rcpf(1.0f + __expf(-x));
}

// bond-order term and its d/dr.  p = (r/r0)^bo2, inv_r = 1/r precomputed.
__device__ __forceinline__ void bop_calc(float r, float inv_r, float bo1, float bo2,
                                         float p, float& f, float& df) {
    const float BOTOL = 0.001f;
    float e    = (1.0f + BOTOL) * __expf(bo1 * p);
    float dedr = e * bo1 * bo2 * p * inv_r;      // d e / d r
    const float rmin = BOTOL, rmax = 2.0f * BOTOL;
    const float rterm = -1.0e9f;                 // 1/(rmin-rmax)^3
    float tp, dtp;
    if (e > rmax)      { tp = 1.0f; dtp = 0.0f; }
    else if (e > rmin) {
        float rd   = rmin - e;
        float trm1 = rmin + 2.0f * e - 3.0f * rmax;
        tp  = rterm * rd * rd * trm1;
        dtp = rterm * (2.0f * rd * rd - 2.0f * rd * trm1);
    } else             { tp = 0.0f; dtp = 0.0f; }
    f  = tp * (e - BOTOL);
    df = (dtp * (e - BOTOL) + tp) * dedr;
}

__global__ void zero_kernel(float* p, int n) {
    int i = blockIdx.x * blockDim.x + threadIdx.x;
    if (i < n) p[i] = 0.0f;
}

__global__ __launch_bounds__(256)
void reaxff_nn_force_kernel(
    const float* __restrict__ x, const float* __restrict__ cell, const float* __restrict__ rcell,
    const float* __restrict__ fm_wi, const float* __restrict__ fm_bi,
    const float* __restrict__ fm_w,  const float* __restrict__ fm_b,
    const float* __restrict__ fm_wo, const float* __restrict__ fm_bo,
    const float* __restrict__ fe_wi, const float* __restrict__ fe_bi,
    const float* __restrict__ fe_w,  const float* __restrict__ fe_b,
    const float* __restrict__ fe_wo, const float* __restrict__ fe_bo,
    const float* __restrict__ desi_p, const int* __restrict__ bdid,
    float* __restrict__ out)   // out[0..15]=E, out[16..]=force (16x768x3)
{
    // ---- per-layer A fragments: A = W^T (M=out feat, K=in feat), bias in row K=15 ----
    __shared__ v16h sA[NLAYER * 32];
    {
        _Float16* sAh = (_Float16*)sA;
        for (int idx = threadIdx.x; idx < NLAYER * 32 * 16; idx += blockDim.x) {
            int l    = idx >> 9;         // layer (512 halves per layer)
            int rem  = idx & 511;
            int lane = rem >> 4;
            int e    = rem & 15;
            const float* W; const float* Bv; int kin, kout;
            if (l == 0)       { W = fm_wi;             Bv = fm_bi;          kin = 3; kout = 9; }
            else if (l <= 5)  { W = fm_w + (l - 1)*81; Bv = fm_b + (l-1)*9; kin = 9; kout = 9; }
            else if (l == 6)  { W = fm_wo;             Bv = fm_bo;          kin = 9; kout = 3; }
            else if (l == 7)  { W = fe_wi;             Bv = fe_bi;          kin = 3; kout = 8; }
            else if (l <= 12) { W = fe_w + (l - 8)*64; Bv = fe_b + (l-8)*8; kin = 8; kout = 8; }
            else              { W = fe_wo;             Bv = fe_bo;          kin = 8; kout = 1; }
            int M = lane & 15;
            int K = ((lane >= 16) ? 16 : 0) + e;  // f16 A 16x32 contiguous k layout
            float v = 0.0f;
            if (M < kout) {
                if (K == 15)      v = Bv[M];            // bias row
                else if (K < kin) v = W[K * kout + M];  // W^T
            }
            sAh[idx] = (_Float16)v;
        }
    }
    __syncthreads();

    const float desi = desi_p[0];
    const int lane   = threadIdx.x & 31;
    const bool lower = lane < 16;
    const int nwaves = (gridDim.x * blockDim.x) >> 5;
    const int wid    = (blockIdx.x * blockDim.x + threadIdx.x) >> 5;

    for (int t = wid; t < TILES_TOTAL; t += nwaves) {
        const int b  = t / TILES_PER_BATCH;
        const int k0 = (t % TILES_PER_BATCH) * 16;
        const float* cB = cell  + b * 9;
        const float* rB = rcell + b * 9;

        // ---------- per-bond scalar part (lanes 0..15, one bond each) ----------
        float vfx = 0.f, vfy = 0.f, vfz = 0.f, inv_r = 1.0f;
        int ai = 0, aj = 0;
        float f0 = 0.f, f1 = 0.f, f2 = 0.f;   // feats
        float d0 = 0.f, d1 = 0.f, d2 = 0.f;   // d feats / dr
        if (lower) {
            int k = k0 + lane;
            ai = bdid[2 * k];
            aj = bdid[2 * k + 1];
            const float* xi = x + (size_t)(b * NATOM + ai) * 3;
            const float* xj = x + (size_t)(b * NATOM + aj) * 3;
            float v0 = xi[0] - xj[0], v1 = xi[1] - xj[1], v2 = xi[2] - xj[2];
            // fractional coords + minimum image
            float w0 = v0 * rB[0] + v1 * rB[3] + v2 * rB[6];
            float w1 = v0 * rB[1] + v1 * rB[4] + v2 * rB[7];
            float w2 = v0 * rB[2] + v1 * rB[5] + v2 * rB[8];
            w0 = (w0 - 0.5f > 0.f) ? w0 - 1.f : w0;  w0 = (w0 + 0.5f < 0.f) ? w0 + 1.f : w0;
            w1 = (w1 - 0.5f > 0.f) ? w1 - 1.f : w1;  w1 = (w1 + 0.5f < 0.f) ? w1 + 1.f : w1;
            w2 = (w2 - 0.5f > 0.f) ? w2 - 1.f : w2;  w2 = (w2 + 0.5f < 0.f) ? w2 + 1.f : w2;
            vfx = w0 * cB[0] + w1 * cB[3] + w2 * cB[6];
            vfy = w0 * cB[1] + w1 * cB[4] + w2 * cB[7];
            vfz = w0 * cB[2] + w1 * cB[5] + w2 * cB[8];
            float d2sq = vfx * vfx + vfy * vfy + vfz * vfz + 1e-8f;
            inv_r   = __builtin_amdgcn_rsqf(d2sq);    // v_rsq_f32
            float r = d2sq * inv_r;                   // r = d2 / sqrt(d2)
            // bond-order features + derivatives
            float qs = r / 1.39f; float q2 = qs * qs;
            bop_calc(r, inv_r, -0.077f, 6.0f, q2 * q2 * q2, f0, d0);
            qs = r / 1.30f; q2 = qs * qs;
            bop_calc(r, inv_r, -0.150f, 5.0f, q2 * q2 * qs, f1, d1);
            qs = r / 1.25f; q2 = qs * qs;
            bop_calc(r, inv_r, -0.300f, 4.0f, q2 * q2, f2, d2);
        }

        // ---------- initial B fragments: H^T (value) and tangent ----------
        // Upper lanes (K=16..31) multiply all-zero A columns, so their content
        // only needs to be finite — no lane masking required anywhere.
        v16h hv, ht;
        const _Float16 h0 = (_Float16)0.0f;
#pragma unroll
        for (int e = 0; e < 16; ++e) { hv[e] = h0; ht[e] = h0; }
        hv[0] = (_Float16)f0; hv[1] = (_Float16)f1; hv[2] = (_Float16)f2;
        hv[15] = (_Float16)1.0f;                 // bias pickup (K=15)
        ht[0] = (_Float16)d0; ht[1] = (_Float16)d1; ht[2] = (_Float16)d2;

        const int src = (lane & 15) + 16;
        const v8f czero = {0.f, 0.f, 0.f, 0.f, 0.f, 0.f, 0.f, 0.f};

        // ---------- 13 fused (value + JVP) sigmoid layers ----------
#pragma unroll
        for (int l = 0; l < 13; ++l) {
            const int kout = (l <= 5) ? 9 : ((l == 6) ? 3 : 8);
            const int nlow = (kout < 8) ? kout : 8;   // valid rows in lower half
            v16h a = sA[l * 32 + lane];
            v8f z  = __builtin_amdgcn_wmma_f32_16x16x32_f16(false, a, false, hv,
                                                            (short)0, czero, false, false);
            v8f zt = __builtin_amdgcn_wmma_f32_16x16x32_f16(false, a, false, ht,
                                                            (short)0, czero, false, false);
            v16h nhv, nht;
#pragma unroll
            for (int e = 0; e < 16; ++e) { nhv[e] = h0; nht[e] = h0; }
            float s0 = 0.f, st0 = 0.f;
#pragma unroll
            for (int v = 0; v < 8; ++v) {
                if (v < nlow) {
                    float sg  = sigmoidf_(z[v]);
                    float sgt = sg * (1.0f - sg) * zt[v];
                    nhv[v] = (_Float16)sg;
                    nht[v] = (_Float16)sgt;
                    if (v == 0) { s0 = sg; st0 = sgt; }
                }
            }
            if (kout > 8) {   // row 8 lives in the upper half: one shuffle pair
                nhv[8] = (_Float16)__shfl(s0,  src, 32);
                nht[8] = (_Float16)__shfl(st0, src, 32);
            }
            nhv[15] = (_Float16)1.0f;   // bias pickup for next layer
            hv = nhv; ht = nht;
        }

        // ---------- final layer (fe_wo: 8 -> 1) ----------
        {
            v16h a = sA[13 * 32 + lane];
            v8f z  = __builtin_amdgcn_wmma_f32_16x16x32_f16(false, a, false, hv,
                                                            (short)0, czero, false, false);
            v8f zt = __builtin_amdgcn_wmma_f32_16x16x32_f16(false, a, false, ht,
                                                            (short)0, czero, false, false);
            float esum = 0.f;
            if (lower) {
                float esi  = sigmoidf_(z[0]);            // row M=0, bond N=lane
                float esit = esi * (1.0f - esi) * zt[0]; // d esi / d r
                esum = -desi * esi;                      // bond energy
                float dEdr = -desi * esit;
                // dr/dx[ai] = rcell * (cell * vr/r)
                float a0 = vfx * inv_r, a1 = vfy * inv_r, a2 = vfz * inv_r;
                float q0 = cB[0] * a0 + cB[1] * a1 + cB[2] * a2;
                float q1 = cB[3] * a0 + cB[4] * a1 + cB[5] * a2;
                float q2 = cB[6] * a0 + cB[7] * a1 + cB[8] * a2;
                float g0 = rB[0] * q0 + rB[1] * q1 + rB[2] * q2;
                float g1 = rB[3] * q0 + rB[4] * q1 + rB[5] * q2;
                float g2 = rB[6] * q0 + rB[7] * q1 + rB[8] * q2;
                float fs = -dEdr;                        // force = -dE/dx
                float* F = out + 16;
                float* Fi = F + (size_t)(b * NATOM + ai) * 3;
                float* Fj = F + (size_t)(b * NATOM + aj) * 3;
                atomicAdd(&Fi[0],  fs * g0);
                atomicAdd(&Fi[1],  fs * g1);
                atomicAdd(&Fi[2],  fs * g2);
                atomicAdd(&Fj[0], -fs * g0);
                atomicAdd(&Fj[1], -fs * g1);
                atomicAdd(&Fj[2], -fs * g2);
            }
            // batch energy reduction over lanes 0..15 (upper lanes contribute 0)
            esum += __shfl_xor(esum, 1, 32);
            esum += __shfl_xor(esum, 2, 32);
            esum += __shfl_xor(esum, 4, 32);
            esum += __shfl_xor(esum, 8, 32);
            if (lane == 0) atomicAdd(&out[b], esum);
        }
    }
}

extern "C" void kernel_launch(void* const* d_in, const int* in_sizes, int n_in,
                              void* d_out, int out_size, void* d_ws, size_t ws_size,
                              hipStream_t stream) {
    const float* x      = (const float*)d_in[0];
    const float* cell   = (const float*)d_in[1];
    const float* rcell  = (const float*)d_in[2];
    const float* fm_wi  = (const float*)d_in[3];
    const float* fm_bi  = (const float*)d_in[4];
    const float* fm_w   = (const float*)d_in[5];
    const float* fm_b   = (const float*)d_in[6];
    const float* fm_wo  = (const float*)d_in[7];
    const float* fm_bo  = (const float*)d_in[8];
    const float* fe_wi  = (const float*)d_in[9];
    const float* fe_bi  = (const float*)d_in[10];
    const float* fe_w   = (const float*)d_in[11];
    const float* fe_b   = (const float*)d_in[12];
    const float* fe_wo  = (const float*)d_in[13];
    const float* fe_bo  = (const float*)d_in[14];
    const float* desi   = (const float*)d_in[15];
    const int*   bdid   = (const int*)d_in[16];
    float* out = (float*)d_out;

    zero_kernel<<<(out_size + 255) / 256, 256, 0, stream>>>(out, out_size);
    reaxff_nn_force_kernel<<<96, 256, 0, stream>>>(
        x, cell, rcell,
        fm_wi, fm_bi, fm_w, fm_b, fm_wo, fm_bo,
        fe_wi, fe_bi, fe_w, fe_b, fe_wo, fe_bo,
        desi, bdid, out);
}
